// MultiHeadedPolarRotateAttention_66657892433936
// MI455X (gfx1250) — matrix operand order repaired and checked by reading, not compiled, not run on hardware
//
#include <hip/hip_runtime.h>
#include <cstdint>

// ---------------------------------------------------------------------------
// MI455X (gfx1250) implementation of MultiHeadedPolarRotateAttention.
// Compute-bound (~275 GFLOP over ~300 MB -> ~900 FLOP/B): all matmuls run on
// v_wmma_f32_16x16x32_bf16. Pipeline:
//   [6x proj GEMM fp32->bf16] -> [flash attention, bf16 WMMA, online softmax,
//   TDM double-buffered K-tile staging] -> [2x out-proj GEMM bf16->fp32]
// Workspace: 8 x 16 MiB bf16 buffers (Qr,Kr,Vr,Qi,Ki,Vi,Xr,Xi) = 128 MiB.
// ---------------------------------------------------------------------------

typedef __attribute__((ext_vector_type(16))) __bf16 v16bf;
typedef __attribute__((ext_vector_type(8)))  __bf16 v8bf;
typedef __attribute__((ext_vector_type(4)))  __bf16 v4bf;
typedef __attribute__((ext_vector_type(8)))  float  v8f;
typedef __attribute__((ext_vector_type(4)))  float  v4f;

#define B_    4
#define S_    2048
#define D_    1024
#define H_    16
#define DK_   64
#define M_TOT (B_ * S_)   // 8192

// ---- Tensor Data Mover availability / toolchain arity dispatch ------------
#if defined(__has_builtin)
#if __has_builtin(__builtin_amdgcn_tensor_load_to_lds) && \
    __has_builtin(__builtin_amdgcn_s_wait_tensorcnt)
#define USE_TDM 1
#else
#define USE_TDM 0
#endif
#else
#define USE_TDM 0
#endif

#if __has_include(<hip/amd_detail/amd_gfx1250_TDM.h>)
#define TDM_6ARG 1   // therock-10.0 headers -> 6-arg builtin
#else
#define TDM_6ARG 0   // ROCm 7.2 -> 5-arg builtin
#endif

__device__ __forceinline__ v8f wmma_bf16(v16bf a, v16bf b, v8f c) {
  // (neg_a, A, neg_b, B, c_mod, C, reuse_a, reuse_b)
  return __builtin_amdgcn_wmma_f32_16x16x32_bf16(false, a, false, b,
                                                 (short)0, c, false, false);
}

__device__ __forceinline__ v16bf cat8(v8bf lo, v8bf hi) {
  return __builtin_shufflevector(lo, hi, 0, 1, 2, 3, 4, 5, 6, 7,
                                 8, 9, 10, 11, 12, 13, 14, 15);
}

__device__ __forceinline__ v8f vzero8() {
  v8f z = {0.f, 0.f, 0.f, 0.f, 0.f, 0.f, 0.f, 0.f};
  return z;
}

#if USE_TDM
typedef __attribute__((ext_vector_type(4))) unsigned int v4u;
typedef __attribute__((ext_vector_type(8))) int          v8i;
typedef __attribute__((ext_vector_type(4))) int          v4i;

// DMA a 32-row x 64-col bf16 tile (row stride 64) from global to LDS.
// D# per CDNA5 ISA ch.8: group0 = {count, lds_addr, global_addr, type=2},
// group1 = {data_size=2B, tensor_dim0=64, tensor_dim1=2048, tile=64x32,
//           tensor_dim0_stride=64}.  Groups 2/3 zero (2D tensor).
__device__ __forceinline__ void tdm_load_tile_32x64(const __bf16* g,
                                                    __bf16* lds) {
  const unsigned long long ga = (unsigned long long)(uintptr_t)g;
  const unsigned int la = (unsigned int)(uintptr_t)lds;  // low 32b = LDS offset
  v4u g0;
  g0[0] = 1u;                                   // count=1 (valid, user mode)
  g0[1] = la;                                   // lds_addr
  g0[2] = (unsigned int)ga;                     // global_addr[31:0]
  g0[3] = (unsigned int)((ga >> 32) & 0x01FFFFFFu) | (2u << 30); // [56:32]|type=2
  v8i g1;
  g1[0] = 0x00010000;                           // data_size = 1 -> 2 bytes
  g1[1] = (int)(64u << 16);                     // tensor_dim0 = 64   (bits 79:48)
  g1[2] = (int)(2048u << 16);                   // tensor_dim1 = 2048 (bits 111:80)
  g1[3] = (int)(64u << 16);                     // tile_dim0 = 64     (bits 127:112)
  g1[4] = 32;                                   // tile_dim1 = 32     (bits 143:128)
  g1[5] = 64;                                   // tensor_dim0_stride (bits 207:160)
  g1[6] = 0;
  g1[7] = 0;
  v4i z4 = {0, 0, 0, 0};
#if TDM_6ARG
  v8i z8 = {0, 0, 0, 0, 0, 0, 0, 0};
  __builtin_amdgcn_tensor_load_to_lds(g0, g1, z4, z4, z8, 0);
#else
  __builtin_amdgcn_tensor_load_to_lds(g0, g1, z4, z4, 0);
#endif
}
#endif  // USE_TDM

// ---------------------------------------------------------------------------
// GEMM: C(MxN) = A(MxK) * W(NxK)^T + bias(N)
//   OUT_MODE 0: write bf16, permuted to (B,H,S,Dk) layout (projection path)
//   OUT_MODE 1: write fp32, row-major MxN (output projection into d_out)
//   A_BF16: A operand is bf16 (workspace) vs fp32 (harness inputs)
// Tiling: 128x128x32, 256 threads = 8 waves in a 2x4 grid, each wave 64x32.
// ---------------------------------------------------------------------------
template <int OUT_MODE, bool A_BF16>
__global__ void __launch_bounds__(256)
gemm_wmma(const void* __restrict__ Ain, const float* __restrict__ W,
          const float* __restrict__ bias, void* __restrict__ Out,
          int M, int N, int K) {
  __shared__ alignas(32) __bf16 sA[128 * 32];
  __shared__ alignas(32) __bf16 sB[128 * 32];

  const int lane = threadIdx.x & 31;
  const int wave = threadIdx.x >> 5;      // 0..7
  const int wm   = wave >> 2;             // 0..1 (64 rows each)
  const int wn   = wave & 3;              // 0..3 (32 cols each)
  const int half = lane >> 4;
  const int ln   = lane & 15;

  const int m0 = blockIdx.x * 128;
  const int n0 = blockIdx.y * 128;

  v8f acc[4][2];
#pragma unroll
  for (int nt = 0; nt < 2; ++nt) {
    float bv = bias[n0 + wn * 32 + nt * 16 + ln];
#pragma unroll
    for (int mt = 0; mt < 4; ++mt) {
#pragma unroll
      for (int e = 0; e < 8; ++e) acc[mt][nt][e] = bv;
    }
  }

  const int tr = threadIdx.x >> 1;         // staging row 0..127
  const int tc = (threadIdx.x & 1) * 16;   // staging col segment

  for (int kk = 0; kk < K; kk += 32) {
#pragma unroll
    for (int i = 0; i < 4; ++i) {
      int c = tc + i * 4;
      v4bf av;
      if constexpr (A_BF16) {
        av = *(const v4bf*)((const __bf16*)Ain + (size_t)(m0 + tr) * K + kk + c);
      } else {
        v4f af = *(const v4f*)((const float*)Ain + (size_t)(m0 + tr) * K + kk + c);
        av[0] = (__bf16)af[0]; av[1] = (__bf16)af[1];
        av[2] = (__bf16)af[2]; av[3] = (__bf16)af[3];
      }
      *(v4bf*)(&sA[tr * 32 + c]) = av;

      v4f wf = *(const v4f*)(W + (size_t)(n0 + tr) * K + kk + c);
      v4bf wv;
      wv[0] = (__bf16)wf[0]; wv[1] = (__bf16)wf[1];
      wv[2] = (__bf16)wf[2]; wv[3] = (__bf16)wf[3];
      *(v4bf*)(&sB[tr * 32 + c]) = wv;
    }
    if (kk + 32 < K) {   // global_prefetch_b8 of next K slab
      __builtin_prefetch((const char*)Ain +
                             ((size_t)(m0 + tr) * K + kk + 32 + tc) * (A_BF16 ? 2 : 4),
                         0, 1);
      __builtin_prefetch(W + (size_t)(n0 + tr) * K + kk + 32 + tc, 0, 1);
    }
    __syncthreads();

    const int kb0 = half ? 16 : 0;  // A fragment K base
    const int kb1 = half ? 16 : 0;  // B fragment K base
    v16bf aF[4], bF[2];
#pragma unroll
    for (int mt = 0; mt < 4; ++mt) {
      int ar = wm * 64 + mt * 16 + ln;
      v8bf lo = *(const v8bf*)(&sA[ar * 32 + kb0]);
      v8bf hi = *(const v8bf*)(&sA[ar * 32 + kb0 + 8]);
      aF[mt] = cat8(lo, hi);
    }
#pragma unroll
    for (int nt = 0; nt < 2; ++nt) {
      int br = wn * 32 + nt * 16 + ln;
      bF[nt] = *(const v16bf*)(&sB[br * 32 + kb1]);
    }
#pragma unroll
    for (int mt = 0; mt < 4; ++mt)
#pragma unroll
      for (int nt = 0; nt < 2; ++nt)
        acc[mt][nt] = wmma_bf16(aF[mt], bF[nt], acc[mt][nt]);

    __syncthreads();
  }

#pragma unroll
  for (int mt = 0; mt < 4; ++mt) {
#pragma unroll
    for (int nt = 0; nt < 2; ++nt) {
      int col = n0 + wn * 32 + nt * 16 + ln;
#pragma unroll
      for (int e = 0; e < 8; ++e) {
        int row = m0 + wm * 64 + mt * 16 + e + 8 * half;
        float v = acc[mt][nt][e];
        if constexpr (OUT_MODE == 0) {
          int b = row >> 11, s = row & (S_ - 1);
          int h = col >> 6, dk = col & (DK_ - 1);
          ((__bf16*)Out)[(((size_t)(b * H_ + h) * S_ + s) * DK_) + dk] = (__bf16)v;
        } else {
          ((float*)Out)[(size_t)row * N + col] = v;
        }
      }
    }
  }
}

// ---------------------------------------------------------------------------
// Flash attention over the complex score (QrKr^T + QiKi^T)/8, softmax, then
// P*Vr and P*Vi.  Grid: (S/64, B*H), 128 threads = 4 waves; each wave owns
// 16 query rows with full Dk=64 real+imag accumulators.  K/Ki tiles are
// DMA'd by the Tensor Data Mover into double buffers one tile ahead of the
// compute; V/Vi tiles are staged (transposed) by the waves.
// ---------------------------------------------------------------------------
__global__ void __launch_bounds__(128)
attn_wmma(const __bf16* __restrict__ Qr, const __bf16* __restrict__ Kr,
          const __bf16* __restrict__ Vr, const __bf16* __restrict__ Qi,
          const __bf16* __restrict__ Ki, const __bf16* __restrict__ Vi,
          __bf16* __restrict__ Xr, __bf16* __restrict__ Xi) {
  __shared__ alignas(32) __bf16 sKr[2][32 * 64];
  __shared__ alignas(32) __bf16 sKi[2][32 * 64];
  __shared__ alignas(32) __bf16 sVrT[64 * 32];
  __shared__ alignas(32) __bf16 sViT[64 * 32];
  __shared__ alignas(32) __bf16 sP[4][16 * 32];

  const int lane = threadIdx.x & 31;
  const int wave = threadIdx.x >> 5;   // 0..3
  const int half = lane >> 4;
  const int ln   = lane & 15;

  const int bh = blockIdx.y;           // b*H + h
  const int q0 = blockIdx.x * 64;
  const size_t base = (size_t)bh * S_ * DK_;
  const int NT = S_ / 32;

  // ---- Q fragments, resident for the whole K loop ----
  const int kb0 = half ? 16 : 0;
  v16bf aQr[2], aQi[2];
  {
    const __bf16* qpr = Qr + base + (size_t)(q0 + wave * 16 + ln) * DK_;
    const __bf16* qpi = Qi + base + (size_t)(q0 + wave * 16 + ln) * DK_;
#pragma unroll
    for (int ks = 0; ks < 2; ++ks) {
      v8bf lo = *(const v8bf*)(qpr + ks * 32 + kb0);
      v8bf hi = *(const v8bf*)(qpr + ks * 32 + kb0 + 8);
      aQr[ks] = cat8(lo, hi);
      lo = *(const v8bf*)(qpi + ks * 32 + kb0);
      hi = *(const v8bf*)(qpi + ks * 32 + kb0 + 8);
      aQi[ks] = cat8(lo, hi);
    }
  }

  v8f accR[4], accI[4];
#pragma unroll
  for (int dt = 0; dt < 4; ++dt) { accR[dt] = vzero8(); accI[dt] = vzero8(); }
  float mrow[8], lrow[8];
#pragma unroll
  for (int e = 0; e < 8; ++e) { mrow[e] = -3.0e38f; lrow[e] = 0.f; }

#if USE_TDM
  if (wave == 0) {  // prologue: DMA tile 0 into buffer 0
    tdm_load_tile_32x64(Kr + base, &sKr[0][0]);
    tdm_load_tile_32x64(Ki + base, &sKi[0][0]);
  }
#endif

  for (int kt = 0; kt < NT; ++kt) {
    __syncthreads();   // everyone done reading previous buffers

#if USE_TDM
    if (wave == 0 && kt + 1 < NT) {  // DMA next K/Ki tile into alternate buffer
      size_t noff = base + (size_t)(kt + 1) * 32 * DK_;
      tdm_load_tile_32x64(Kr + noff, &sKr[(kt + 1) & 1][0]);
      tdm_load_tile_32x64(Ki + noff, &sKi[(kt + 1) & 1][0]);
    }
#else
    {  // manual K staging fallback
      int r = threadIdx.x >> 2, cg = (threadIdx.x & 3) * 16;
      size_t off = base + (size_t)(kt * 32 + r) * DK_ + cg;
      *(v8bf*)(&sKr[0][r * 64 + cg])     = *(const v8bf*)(Kr + off);
      *(v8bf*)(&sKr[0][r * 64 + cg + 8]) = *(const v8bf*)(Kr + off + 8);
      *(v8bf*)(&sKi[0][r * 64 + cg])     = *(const v8bf*)(Ki + off);
      *(v8bf*)(&sKi[0][r * 64 + cg + 8]) = *(const v8bf*)(Ki + off + 8);
    }
#endif

    // ---- stage V tiles transposed (Dk-major: sVT[d][k]) ----
    {
      int kl = threadIdx.x & 31, dg = (threadIdx.x >> 5) * 16;
      size_t off = base + (size_t)(kt * 32 + kl) * DK_ + dg;
      v8bf a0 = *(const v8bf*)(Vr + off), a1 = *(const v8bf*)(Vr + off + 8);
      v8bf c0 = *(const v8bf*)(Vi + off), c1 = *(const v8bf*)(Vi + off + 8);
#pragma unroll
      for (int j = 0; j < 8; ++j) {
        sVrT[(dg + j) * 32 + kl]     = a0[j];
        sVrT[(dg + 8 + j) * 32 + kl] = a1[j];
        sViT[(dg + j) * 32 + kl]     = c0[j];
        sViT[(dg + 8 + j) * 32 + kl] = c1[j];
      }
    }

#if USE_TDM
    if (wave == 0) {
      // TDM is in-order per wave: <=2 outstanding means tile kt's pair landed.
      if (kt + 1 < NT) __builtin_amdgcn_s_wait_tensorcnt(2);
      else             __builtin_amdgcn_s_wait_tensorcnt(0);
    }
    const __bf16* kbufR = &sKr[kt & 1][0];
    const __bf16* kbufI = &sKi[kt & 1][0];
#else
    const __bf16* kbufR = &sKr[0][0];
    const __bf16* kbufI = &sKi[0][0];
#endif
    __syncthreads();

    // ---- S tile: 16 q-rows x 32 keys ----
    v8f sc[2] = {vzero8(), vzero8()};
    const int kb1 = half ? 16 : 0;
#pragma unroll
    for (int nt = 0; nt < 2; ++nt) {
      int kr = nt * 16 + ln;
#pragma unroll
      for (int ks = 0; ks < 2; ++ks) {
        v16bf bKr = *(const v16bf*)(&kbufR[kr * 64 + ks * 32 + kb1]);
        sc[nt] = wmma_bf16(aQr[ks], bKr, sc[nt]);
        v16bf bKi = *(const v16bf*)(&kbufI[kr * 64 + ks * 32 + kb1]);
        sc[nt] = wmma_bf16(aQi[ks], bKi, sc[nt]);
      }
    }

    // ---- online softmax (row stats live in width-16 lane groups) ----
    float esc[8];
#pragma unroll
    for (int e = 0; e < 8; ++e) {
      sc[0][e] *= 0.125f;  // 1/sqrt(Dk)
      sc[1][e] *= 0.125f;
      float v = fmaxf(sc[0][e], sc[1][e]);
#pragma unroll
      for (int off = 8; off > 0; off >>= 1) v = fmaxf(v, __shfl_xor(v, off, 16));
      float mn = fmaxf(mrow[e], v);
      esc[e] = __expf(mrow[e] - mn);
      mrow[e] = mn;
      sc[0][e] = __expf(sc[0][e] - mn);
      sc[1][e] = __expf(sc[1][e] - mn);
      float rs = sc[0][e] + sc[1][e];
#pragma unroll
      for (int off = 8; off > 0; off >>= 1) rs += __shfl_xor(rs, off, 16);
      lrow[e] = lrow[e] * esc[e] + rs;
    }
#pragma unroll
    for (int dt = 0; dt < 4; ++dt)
#pragma unroll
      for (int e = 0; e < 8; ++e) { accR[dt][e] *= esc[e]; accI[dt][e] *= esc[e]; }

    // ---- P: C-layout -> A-fragment via per-wave LDS (in-order within wave) --
    __bf16* pw = &sP[wave][0];
#pragma unroll
    for (int nt = 0; nt < 2; ++nt)
#pragma unroll
      for (int e = 0; e < 8; ++e)
        pw[(e + 8 * half) * 32 + nt * 16 + ln] = (__bf16)sc[nt][e];

    v8bf plo = *(const v8bf*)(&pw[ln * 32 + kb0]);
    v8bf phi = *(const v8bf*)(&pw[ln * 32 + kb0 + 8]);
    v16bf aP = cat8(plo, phi);

    // ---- P @ V (real and imag) ----
#pragma unroll
    for (int dt = 0; dt < 4; ++dt) {
      int dr = dt * 16 + ln;
      v16bf bVr = *(const v16bf*)(&sVrT[dr * 32 + kb1]);
      accR[dt] = wmma_bf16(aP, bVr, accR[dt]);
      v16bf bVi = *(const v16bf*)(&sViT[dr * 32 + kb1]);
      accI[dt] = wmma_bf16(aP, bVi, accI[dt]);
    }
  }

  // ---- normalize and write X in (B,S,D) bf16 layout ----
  const int b = bh >> 4, h = bh & 15;
#pragma unroll
  for (int e = 0; e < 8; ++e) {
    float inv = 1.0f / lrow[e];
    int s = q0 + wave * 16 + e + 8 * half;
    size_t rb = ((size_t)(b * S_ + s)) * D_ + h * DK_;
#pragma unroll
    for (int dt = 0; dt < 4; ++dt) {
      Xr[rb + dt * 16 + ln] = (__bf16)(accR[dt][e] * inv);
      Xi[rb + dt * 16 + ln] = (__bf16)(accI[dt][e] * inv);
    }
  }
}

// ---------------------------------------------------------------------------
extern "C" void kernel_launch(void* const* d_in, const int* in_sizes, int n_in,
                              void* d_out, int out_size, void* d_ws, size_t ws_size,
                              hipStream_t stream) {
  (void)in_sizes; (void)n_in; (void)out_size; (void)ws_size;

  const float* real_q = (const float*)d_in[0];
  const float* img_q  = (const float*)d_in[1];
  const float* real_k = (const float*)d_in[2];
  const float* img_k  = (const float*)d_in[3];
  const float* real_v = (const float*)d_in[4];
  const float* img_v  = (const float*)d_in[5];
  const float* w_r  = (const float*)d_in[6];
  const float* b_r  = (const float*)d_in[7];
  const float* w_i  = (const float*)d_in[8];
  const float* b_i  = (const float*)d_in[9];
  const float* w_or = (const float*)d_in[10];
  const float* b_or = (const float*)d_in[11];
  const float* w_oi = (const float*)d_in[12];
  const float* b_oi = (const float*)d_in[13];
  float* out = (float*)d_out;

  const size_t PLANE = (size_t)M_TOT * D_;
  char* ws = (char*)d_ws;
  const size_t SZ = PLANE * sizeof(__bf16);   // 16 MiB each
  __bf16* Qr = (__bf16*)(ws + 0 * SZ);
  __bf16* Kr = (__bf16*)(ws + 1 * SZ);
  __bf16* Vr = (__bf16*)(ws + 2 * SZ);
  __bf16* Qi = (__bf16*)(ws + 3 * SZ);
  __bf16* Ki = (__bf16*)(ws + 4 * SZ);
  __bf16* Vi = (__bf16*)(ws + 5 * SZ);
  __bf16* Xr = (__bf16*)(ws + 6 * SZ);
  __bf16* Xi = (__bf16*)(ws + 7 * SZ);

  dim3 gG(M_TOT / 128, D_ / 128);   // 64 x 8
  const size_t DD = (size_t)D_ * D_;

  // 6 input projections (fp32 A, bf16 out in (B,H,S,Dk))
  gemm_wmma<0, false><<<gG, 256, 0, stream>>>(real_q, w_r + 0 * DD, b_r + 0 * D_, Qr, M_TOT, D_, D_);
  gemm_wmma<0, false><<<gG, 256, 0, stream>>>(real_k, w_r + 1 * DD, b_r + 1 * D_, Kr, M_TOT, D_, D_);
  gemm_wmma<0, false><<<gG, 256, 0, stream>>>(real_v, w_r + 2 * DD, b_r + 2 * D_, Vr, M_TOT, D_, D_);
  gemm_wmma<0, false><<<gG, 256, 0, stream>>>(img_q,  w_i + 0 * DD, b_i + 0 * D_, Qi, M_TOT, D_, D_);
  gemm_wmma<0, false><<<gG, 256, 0, stream>>>(img_k,  w_i + 1 * DD, b_i + 1 * D_, Ki, M_TOT, D_, D_);
  gemm_wmma<0, false><<<gG, 256, 0, stream>>>(img_v,  w_i + 2 * DD, b_i + 2 * D_, Vi, M_TOT, D_, D_);

  // attention: grid (q-tiles, B*H)
  attn_wmma<<<dim3(S_ / 64, B_ * H_), 128, 0, stream>>>(Qr, Kr, Vr, Qi, Ki, Vi, Xr, Xi);

  // 2 output projections (bf16 A, fp32 out; tuple concatenated in d_out)
  gemm_wmma<1, true><<<gG, 256, 0, stream>>>(Xr, w_or, b_or, out,         M_TOT, D_, D_);
  gemm_wmma<1, true><<<gG, 256, 0, stream>>>(Xi, w_oi, b_oi, out + PLANE, M_TOT, D_, D_);
}
